// CloudShieldHGT_67808943669994
// MI455X (gfx1250) — matrix operand order repaired and checked by reading, not compiled, not run on hardware
//
#include <hip/hip_runtime.h>
#include <hip/hip_bf16.h>
#include <math.h>

// ---------------------------------------------------------------------------
// CDNA5 (gfx1250) HGT forward. wave32.
// Dense 128-K GEMMs: bf16 WMMA 16x16x32 (f32 accum), weights staged once per
// workgroup into LDS as pre-packed B-fragments (ds_load_b128 path), each wave
// computes a 16 x No row-stripe (8 WMMAs back-to-back per K-step for No=128).
// Edge softmax/scatter via hardware f32/u32 atomics.
// ---------------------------------------------------------------------------

typedef __bf16 bf16_t;
typedef bf16_t v16bf __attribute__((ext_vector_type(16)));
typedef float  v8f   __attribute__((ext_vector_type(8)));

#define NPOLICY 100000
#define NUSER   100000
#define NROLE   50000
#define NEDGE   500000
#define HIDDIM  128
#define KDIM    128

// ------------------------------ fills --------------------------------------
__global__ void fill_f32_kernel(float* __restrict__ p, float v, long long n) {
  long long i = (long long)blockIdx.x * blockDim.x + threadIdx.x;
  if (i < n) p[i] = v;
}
__global__ void fill_u32_kernel(unsigned* __restrict__ p, unsigned v, long long n) {
  long long i = (long long)blockIdx.x * blockDim.x + threadIdx.x;
  if (i < n) p[i] = v;
}

// ------------------- small-K dense linear (VALU path) ----------------------
__global__ void linear_small_kernel(const float* __restrict__ X, const float* __restrict__ W,
                                    const float* __restrict__ B, float* __restrict__ Y,
                                    int N, int K, int No, int relu) {
  long long idx = (long long)blockIdx.x * blockDim.x + threadIdx.x;
  if (idx >= (long long)N * No) return;
  int n = (int)(idx / No), o = (int)(idx % No);
  const float* x = X + (long long)n * K;
  float acc = B[o];
  for (int k = 0; k < K; ++k) acc += x[k] * W[(long long)k * No + o];
  if (relu && acc < 0.f) acc = 0.f;
  Y[idx] = acc;
}

// ------------------------- WMMA bf16 GEMM ----------------------------------
// Y[N,No] = X[N,128] @ W[128,No] (+ bias), No = NT*16.
// Workgroup = 256 threads = 8 waves; covers 128 rows (8 row-tiles).
// Stage 1: cooperatively pack W into LDS as bf16 B-fragments so each lane's
//   v16bf fragment is 32 contiguous bytes (ds_load_b128 x2).
//   Packed index: (((kb*NT + nt)*32 + lane)*16 + i) ,
//   mapping  k = kb*32 + (i>>3)*16 + (lane>>4)*8 + (i&7) , n = nt*16 + (lane&15).
// Stage 2: per wave, per K-step: one A fragment (4x global_load_b128, cvt to
//   bf16) reused across NT consecutive WMMAs.
// A layout (ISA 7.12.2, 16-bit 16x32): lane half=lane>>4, m=lane&15;
//   elem i<8: K=half*8+i ; elem i>=8: K=16+half*8+(i-8).
// C/D: c[r] = D[m0 + r + 8*half][n0 + (lane&15)].
template <int NT>
__global__ void wmma_gemm_kernel(const float* __restrict__ X, const float* __restrict__ W,
                                 const float* __restrict__ B, float* __restrict__ Y, int N) {
  constexpr int No = NT * 16;
  __shared__ __align__(32) bf16_t wpk[KDIM * No];

  // ---- cooperative W pack (all 256 threads; before any wave-guard exit) ----
  for (int idx = threadIdx.x; idx < KDIM * No; idx += 256) {
    int i     = idx & 15;
    int lanep = (idx >> 4) & 31;
    int t2    = idx >> 9;          // kb*NT + nt
    int nt    = t2 & (NT - 1);
    int kb    = t2 / NT;
    int k = kb * 32 + (i >> 3) * 16 + (lanep >> 4) * 8 + (i & 7);
    int n = nt * 16 + (lanep & 15);
    wpk[idx] = (bf16_t)W[(long long)k * No + n];
  }
  __syncthreads();

  int wave = threadIdx.x >> 5;
  int lane = threadIdx.x & 31;
  int mt = blockIdx.x * 8 + wave;            // 16-row tile index
  if (mt >= (N >> 4)) return;                // wave-uniform guard
  int half = lane >> 4, l = lane & 15;

  const float* xrow = X + (long long)(mt * 16 + l) * KDIM;
  v8f acc[NT];
#pragma unroll
  for (int nt = 0; nt < NT; ++nt) acc[nt] = (v8f){};

  for (int kb = 0; kb < KDIM / 32; ++kb) {
    const float4* xp = (const float4*)(xrow + kb * 32);
    float4 a0 = xp[half * 2];
    float4 a1 = xp[half * 2 + 1];
    float4 a2 = xp[4 + half * 2];
    float4 a3 = xp[5 + half * 2];
    v16bf a;
    a[0] = (bf16_t)a0.x;  a[1] = (bf16_t)a0.y;  a[2] = (bf16_t)a0.z;  a[3] = (bf16_t)a0.w;
    a[4] = (bf16_t)a1.x;  a[5] = (bf16_t)a1.y;  a[6] = (bf16_t)a1.z;  a[7] = (bf16_t)a1.w;
    a[8] = (bf16_t)a2.x;  a[9] = (bf16_t)a2.y;  a[10] = (bf16_t)a2.z; a[11] = (bf16_t)a2.w;
    a[12] = (bf16_t)a3.x; a[13] = (bf16_t)a3.y; a[14] = (bf16_t)a3.z; a[15] = (bf16_t)a3.w;
#pragma unroll
    for (int nt = 0; nt < NT; ++nt) {
      const v16bf* bp = (const v16bf*)&wpk[(((kb * NT + nt) * 32) + lane) * 16];
      acc[nt] = __builtin_amdgcn_wmma_f32_16x16x32_bf16(false, a, false, *bp, (short)0,
                                                        acc[nt], false, false);
    }
  }

#pragma unroll
  for (int nt = 0; nt < NT; ++nt) {
    float bias = B ? B[nt * 16 + l] : 0.f;
#pragma unroll
    for (int r = 0; r < 8; ++r)
      Y[(long long)(mt * 16 + r + half * 8) * No + (nt * 16 + l)] = acc[nt][r] + bias;
  }
}

// ---- build block-diagonal 128x128 from rel[4,32,32] (einsum nhd,hde->nhe) --
__global__ void blockdiag_kernel(const float* __restrict__ rel, float* __restrict__ bd) {
  int idx = blockIdx.x * blockDim.x + threadIdx.x;   // 128*128
  if (idx >= HIDDIM * HIDDIM) return;
  int i = idx >> 7, j = idx & 127;
  int hi = i >> 5, hj = j >> 5;
  bd[idx] = (hi == hj) ? rel[hi * 1024 + (i & 31) * 32 + (j & 31)] : 0.f;
}

// ------------------------------ elementwise --------------------------------
__global__ void gelu_kernel(float* __restrict__ x, long long n) {
  long long i = (long long)blockIdx.x * blockDim.x + threadIdx.x;
  if (i >= n) return;
  float v = x[i];
  x[i] = 0.5f * v * (1.f + erff(v * 0.7071067811865476f));
}
__global__ void relu_kernel(float* __restrict__ x, long long n) {
  long long i = (long long)blockIdx.x * blockDim.x + threadIdx.x;
  if (i >= n) return;
  float v = x[i];
  x[i] = v > 0.f ? v : 0.f;
}
// hnew = relu( a*o + (1-a)*hprev ),  a = sigmoid(skip[0])
__global__ void blend_relu_kernel(const float* __restrict__ o, const float* __restrict__ hprev,
                                  const float* __restrict__ skip, float* __restrict__ hnew,
                                  long long n) {
  long long i = (long long)blockIdx.x * blockDim.x + threadIdx.x;
  if (i >= n) return;
  float a = 1.f / (1.f + expf(-skip[0]));
  float v = a * o[i] + (1.f - a) * hprev[i];
  hnew[i] = v > 0.f ? v : 0.f;
}

// ------------------------------ edge phases --------------------------------
__device__ __forceinline__ unsigned f2ord(float f) {
  unsigned b = __float_as_uint(f);
  return (b & 0x80000000u) ? ~b : (b | 0x80000000u);
}
__device__ __forceinline__ float ord2f(unsigned u) {
  return (u & 0x80000000u) ? __uint_as_float(u & 0x7FFFFFFFu) : __uint_as_float(~u);
}

// logit[e,h] = (sum_d krel[src,h*32+d]*q[dst,h*32+d]) * p_rel[h] / sqrt(32)
__global__ void edge_logits_kernel(const float* __restrict__ krel, const float* __restrict__ q,
                                   const int* __restrict__ src, const int* __restrict__ dst,
                                   const float* __restrict__ prel, float* __restrict__ logit,
                                   unsigned* __restrict__ mu, int E) {
  int idx = blockIdx.x * blockDim.x + threadIdx.x;
  if (idx >= E * 4) return;
  int e = idx >> 2, h = idx & 3;
  int s = src[e], d = dst[e];
  const float* kp = krel + (long long)s * HIDDIM + h * 32;
  const float* qp = q    + (long long)d * HIDDIM + h * 32;
  float acc = 0.f;
#pragma unroll
  for (int i = 0; i < 32; ++i) acc += kp[i] * qp[i];
  float lg = acc * prel[h] * 0.17677669529663687f; // 1/sqrt(32)
  logit[idx] = lg;
  atomicMax(mu + (long long)d * 4 + h, f2ord(lg));
}

// ex = exp(logit - m[dst]); den[dst] += ex  (logit buffer overwritten with ex)
__global__ void edge_expsum_kernel(float* __restrict__ logit, const int* __restrict__ dst,
                                   const unsigned* __restrict__ mu, float* __restrict__ den,
                                   int E) {
  int idx = blockIdx.x * blockDim.x + threadIdx.x;
  if (idx >= E * 4) return;
  int e = idx >> 2, h = idx & 3;
  int d = dst[e];
  float m = ord2f(mu[(long long)d * 4 + h]);
  float ex = expf(logit[idx] - m);
  logit[idx] = ex;
  atomicAdd(den + (long long)d * 4 + h, ex);
}

// out[dst, f] += vrel[src, f] * ex[e, f/32] / max(den,1)
__global__ void edge_scatter_kernel(const float* __restrict__ vrel, const float* __restrict__ ex,
                                    const float* __restrict__ den, const int* __restrict__ src,
                                    const int* __restrict__ dst, float* __restrict__ out, int E) {
  long long idx = (long long)blockIdx.x * blockDim.x + threadIdx.x;
  if (idx >= (long long)E * HIDDIM) return;
  int e = (int)(idx >> 7);
  int f = (int)(idx & 127);
  int h = f >> 5;
  int s = src[e], d = dst[e];
  float dn = den[(long long)d * 4 + h];
  float alpha = ex[(long long)e * 4 + h] / (dn > 0.f ? dn : 1.f);
  atomicAdd(out + (long long)d * HIDDIM + f, vrel[(long long)s * HIDDIM + f] * alpha);
}

// ---------------------------------------------------------------------------
extern "C" void kernel_launch(void* const* d_in, const int* in_sizes, int n_in,
                              void* d_out, int out_size, void* d_ws, size_t ws_size,
                              hipStream_t stream) {
  (void)in_sizes; (void)n_in; (void)out_size; (void)ws_size;
  const int NP = NPOLICY, NU = NUSER, NR = NROLE, E = NEDGE;

  // ------------ input pointers (setup_inputs dict order; params = jax pytree
  // flatten: sorted dict keys, list order for layers) ------------------------
  const float* x_policy = (const float*)d_in[0];
  const float* x_user   = (const float*)d_in[1];
  const float* x_role   = (const float*)d_in[2];
  const int* eptr[4] = { (const int*)d_in[3],   // user__attached_to__policy
                         (const int*)d_in[4],   // role__has__policy
                         (const int*)d_in[5],   // policy__applies_to__role
                         (const int*)d_in[6] }; // user__assumes__role
  const float* cls1_b = (const float*)d_in[7];
  const float* cls1_w = (const float*)d_in[8];
  const float* cls2_b = (const float*)d_in[9];
  const float* cls2_w = (const float*)d_in[10];
  // params["lin"] leaves at the tail (key "lin" sorts after "layers")
  const float* lin_b[3] = { (const float*)d_in[89], (const float*)d_in[93], (const float*)d_in[91] }; // policy,user,role
  const float* lin_w[3] = { (const float*)d_in[90], (const float*)d_in[94], (const float*)d_in[92] };

  // ------------------------- workspace carve -------------------------------
  float* wp = (float*)d_ws;
  auto alloc = [&](long long n) { float* p = wp; wp += n; return p; };
  const long long PL = (long long)NP * HIDDIM, UL = (long long)NU * HIDDIM, RL = (long long)NR * HIDDIM;
  float* hp[2] = { alloc(PL), alloc(PL) };   // policy ping-pong
  float* hr[2] = { alloc(RL), alloc(RL) };   // role ping-pong
  float* hu    = alloc(UL);                  // user (never updated by conv)
  float* kb[3] = { alloc(PL), alloc(UL), alloc(RL) };  // k per type (policy,user,role)
  float* vb[3] = { alloc(PL), alloc(UL), alloc(RL) };
  float* q_p = alloc(PL);
  float* q_r = alloc(RL);
  float* krel = alloc(PL);                   // per-edge-type scratch (max src = 100K)
  float* vrel = alloc(PL);
  float* out_p = alloc(PL);
  float* out_r = alloc(RL);
  float* oproj = alloc(PL);
  float* logitb = alloc((long long)E * 4);
  unsigned* mu = (unsigned*)alloc((long long)NP * 4);
  float* den = alloc((long long)NP * 4);
  float* bd  = alloc(HIDDIM * HIDDIM);
  float* clsh = alloc((long long)NP * 64);

  // --------------------------- launch helpers ------------------------------
  auto fillf = [&](float* p, float v, long long n) {
    fill_f32_kernel<<<(unsigned)((n + 255) / 256), 256, 0, stream>>>(p, v, n);
  };
  auto fillu = [&](unsigned* p, unsigned v, long long n) {
    fill_u32_kernel<<<(unsigned)((n + 255) / 256), 256, 0, stream>>>(p, v, n);
  };
  auto gemm = [&](const float* X, const float* W, const float* B, float* Y,
                  int N, int No) {
    unsigned blocks = (unsigned)(((N >> 4) + 7) / 8);
    if (No == 128)
      wmma_gemm_kernel<8><<<blocks, 256, 0, stream>>>(X, W, B, Y, N);
    else
      wmma_gemm_kernel<4><<<blocks, 256, 0, stream>>>(X, W, B, Y, N);
  };
  auto linsmall = [&](const float* X, const float* W, const float* B, float* Y,
                      int N, int K, int No, int relu) {
    long long t = (long long)N * No;
    linear_small_kernel<<<(unsigned)((t + 255) / 256), 256, 0, stream>>>(X, W, B, Y, N, K, No, relu);
  };

  // ------------------- stage 0: input linear + relu ------------------------
  linsmall(x_policy, lin_w[0], lin_b[0], hp[0], NP, 16, HIDDIM, 1);
  linsmall(x_user,   lin_w[1], lin_b[1], hu,    NU, 12, HIDDIM, 1);
  linsmall(x_role,   lin_w[2], lin_b[2], hr[0], NR, 8,  HIDDIM, 1);

  // edge-type metadata: {uap, rhp, par, uar}
  const int relSorted[4] = { 3, 1, 0, 2 };     // index in sorted a_rel/m_rel/p_rel
  const int srcT[4] = { 1, 2, 0, 1 };          // 0 policy, 1 user, 2 role
  const int srcN[4] = { NU, NR, NP, NU };
  const int dstPol[4] = { 1, 1, 0, 0 };        // 1 -> target policy, 0 -> target role

  int cur = 0;
  for (int layer = 0; layer < 2; ++layer) {
    const int base = 11 + 39 * layer;
    auto lf = [&](int i) { return (const float*)d_in[base + i]; };
    const float* hcur[3] = { hp[cur], hu, hr[cur] };

    // k/q/v projections (HID x HID GEMMs, WMMA)
    gemm(hcur[0], lf(11), lf(10), kb[0], NP, HIDDIM);
    gemm(hcur[2], lf(13), lf(12), kb[2], NR, HIDDIM);
    gemm(hcur[1], lf(15), lf(14), kb[1], NU, HIDDIM);
    gemm(hcur[0], lf(34), lf(33), vb[0], NP, HIDDIM);
    gemm(hcur[2], lf(36), lf(35), vb[2], NR, HIDDIM);
    gemm(hcur[1], lf(38), lf(37), vb[1], NU, HIDDIM);
    gemm(hcur[0], lf(25), lf(24), q_p, NP, HIDDIM);
    gemm(hcur[2], lf(27), lf(26), q_r, NR, HIDDIM);

    fillf(out_p, 0.f, PL);
    fillf(out_r, 0.f, RL);

    for (int t = 0; t < 4; ++t) {
      int rs = relSorted[t];
      // relation transforms as block-diagonal WMMA GEMMs
      blockdiag_kernel<<<64, 256, 0, stream>>>(lf(6 + rs), bd);
      gemm(kb[srcT[t]], bd, nullptr, krel, srcN[t], HIDDIM);
      blockdiag_kernel<<<64, 256, 0, stream>>>(lf(16 + rs), bd);
      gemm(vb[srcT[t]], bd, nullptr, vrel, srcN[t], HIDDIM);

      int Nt = dstPol[t] ? NP : NR;
      float* outb = dstPol[t] ? out_p : out_r;
      const float* qb = dstPol[t] ? q_p : q_r;
      const int* src = eptr[t];
      const int* dst = eptr[t] + E;
      fillu(mu, 0u, (long long)Nt * 4);
      fillf(den, 0.f, (long long)Nt * 4);
      edge_logits_kernel<<<(E * 4 + 255) / 256, 256, 0, stream>>>(krel, qb, src, dst,
                                                                  lf(20 + rs), logitb, mu, E);
      edge_expsum_kernel<<<(E * 4 + 255) / 256, 256, 0, stream>>>(logitb, dst, mu, den, E);
      edge_scatter_kernel<<<(unsigned)(((long long)E * HIDDIM + 255) / 256), 256, 0, stream>>>(
          vrel, logitb, den, src, dst, outb, E);
    }

    // post: gelu -> a-proj (WMMA) -> skip blend -> relu
    gelu_kernel<<<(unsigned)((PL + 255) / 256), 256, 0, stream>>>(out_p, PL);
    gemm(out_p, lf(1), lf(0), oproj, NP, HIDDIM);
    blend_relu_kernel<<<(unsigned)((PL + 255) / 256), 256, 0, stream>>>(oproj, hp[cur], lf(30),
                                                                        hp[cur ^ 1], PL);
    gelu_kernel<<<(unsigned)((RL + 255) / 256), 256, 0, stream>>>(out_r, RL);
    gemm(out_r, lf(3), lf(2), oproj, NR, HIDDIM);
    blend_relu_kernel<<<(unsigned)((RL + 255) / 256), 256, 0, stream>>>(oproj, hr[cur], lf(31),
                                                                        hr[cur ^ 1], RL);
    cur ^= 1;
  }

  // ----------------------------- classifier --------------------------------
  gemm(hp[cur], cls1_w, cls1_b, clsh, NP, 64);
  relu_kernel<<<(unsigned)(((long long)NP * 64 + 255) / 256), 256, 0, stream>>>(clsh, (long long)NP * 64);
  linsmall(clsh, cls2_w, cls2_b, (float*)d_out, NP, 64, 3, 0);

  // emb output = final policy h
  hipMemcpyAsync((float*)d_out + (long long)NP * 3, hp[cur], (size_t)PL * sizeof(float),
                 hipMemcpyDeviceToDevice, stream);
}
